// TransformerSimples_1606317769341
// MI455X (gfx1250) — compile-verified, matrix operand order checked by reading, not run on hardware
//
#include <hip/hip_runtime.h>
#include <hip/hip_bf16.h>

// ---------------------------------------------------------------------------
// CDNA5 (gfx1250, wave32) WMMA types
// ---------------------------------------------------------------------------
typedef __attribute__((ext_vector_type(16))) __bf16 v16bf;
typedef __attribute__((ext_vector_type(8)))  float  v8f;

union FragAB { unsigned int u[8]; v16bf bf; };
static_assert(sizeof(FragAB) == 32, "frag size");
union U4 { uint4 v; unsigned short s[8]; };

__device__ __forceinline__ unsigned short f32_to_bf16_rne(float f) {
    unsigned int u = __float_as_uint(f);
    return (unsigned short)((u + 0x7FFFu + ((u >> 16) & 1u)) >> 16);
}

// ---------------------------------------------------------------------------
// bf16 WMMA GEMM:  C[z] = alpha * A[z] * op(B[z]) + bias
//   A: [M,K] bf16 row-major
//   TRANSB==1 -> B [N,K] bf16 row-major (weightsT / K for QK^T): fast staging
//   TRANSB==0 -> B [K,N] bf16 row-major (V for PV): LDS-transpose staging
//   OUTF32: 1 -> f32 C, 0 -> bf16 C
//   Block tile 128x128, K-step 32, 256 threads = 8 waves (4 m-groups x 2 n-groups).
//   Each wave: 2 A-frags x 4 B-frags -> 8 WMMAs per K-step.
//   Double-buffered LDS: ONE barrier per K-step; next tile's global loads and
//   LDS stores overlap current tile's WMMA work.
// ---------------------------------------------------------------------------
template <int TRANSB, int OUTF32>
__global__ __launch_bounds__(256)
void gemm_bf16_wmma(const unsigned short* __restrict__ Aptr,
                    const unsigned short* __restrict__ Bptr,
                    const float* __restrict__ bias,
                    void* __restrict__ Cptr,
                    int Kd, int lda, int ldb, int ldc,
                    long long sA, long long sB, long long sC,
                    float alpha) {
    __shared__ alignas(16) unsigned short lds_a[2][128][32];   // [buf][m][k]
    __shared__ alignas(16) unsigned short lds_b[2][128][32];   // [buf][n][k]

    const int tid  = threadIdx.x;
    const int wave = tid >> 5, lane = tid & 31;
    const int half = lane >> 4, l16 = lane & 15;
    const int wm = wave & 3, wn = wave >> 2;

    const int m0 = blockIdx.y * 128, n0 = blockIdx.x * 128;
    const int z  = blockIdx.z;

    const unsigned short* Ab = Aptr + (size_t)z * sA;
    const unsigned short* Bb = Bptr + (size_t)z * sB;

    // --- staging pointers (advance one K-step = 64B = 4 uint4 per iter) ---
    const uint4* pa0 = (const uint4*)(Ab + (size_t)(m0 +      (tid >> 2)) * lda) + (tid & 3);
    const uint4* pa1 = (const uint4*)(Ab + (size_t)(m0 + 64 + (tid >> 2)) * lda) + (tid & 3);
    const uint4* pb0; const uint4* pb1; size_t bstep;
    if constexpr (TRANSB) {   // B [N,K]: contiguous rows, same pattern as A
        pb0 = (const uint4*)(Bb + (size_t)(n0 +      (tid >> 2)) * ldb) + (tid & 3);
        pb1 = (const uint4*)(Bb + (size_t)(n0 + 64 + (tid >> 2)) * ldb) + (tid & 3);
        bstep = 4;
    } else {                  // B [K,N]: read 32x128 tile coalesced, transpose into LDS
        pb0 = (const uint4*)(Bb + (size_t)(     (tid >> 4)) * ldb + n0) + (tid & 15);
        pb1 = (const uint4*)(Bb + (size_t)(16 + (tid >> 4)) * ldb + n0) + (tid & 15);
        bstep = (size_t)4 * ldb;   // 32 rows * ldb elems, in uint4 units
    }

    v8f acc[2][4];
#pragma unroll
    for (int mi = 0; mi < 2; ++mi)
#pragma unroll
        for (int t = 0; t < 4; ++t)
            acc[mi][t] = (v8f){0.f, 0.f, 0.f, 0.f, 0.f, 0.f, 0.f, 0.f};

    uint4 ra0, ra1; U4 rb0, rb1;

    auto loadRegs = [&]() {
        ra0 = *pa0; ra1 = *pa1; rb0.v = *pb0; rb1.v = *pb1;
        pa0 += 4; pa1 += 4; pb0 += bstep; pb1 += bstep;
    };
    auto storeLds = [&](int buf) {
        ((uint4*)lds_a[buf])[tid]       = ra0;
        ((uint4*)lds_a[buf])[tid + 256] = ra1;
        if constexpr (TRANSB) {
            ((uint4*)lds_b[buf])[tid]       = rb0.v;
            ((uint4*)lds_b[buf])[tid + 256] = rb1.v;
        } else {
            const int kr = tid >> 4, nb = (tid & 15) * 8;
#pragma unroll
            for (int e = 0; e < 8; ++e) {
                lds_b[buf][nb + e][kr]      = rb0.s[e];
                lds_b[buf][nb + e][kr + 16] = rb1.s[e];
            }
        }
    };
    // ISA 16-bit A/B fragment layout: lane = row, VGPR v & lane-half select K
    auto loadFrag = [&](const unsigned short (*L)[32], int row, FragAB& f) {
#pragma unroll
        for (int v = 0; v < 8; ++v) {
            int kb = ((v < 4) ? 0 : 16) + half * 8 + (v & 3) * 2;
            f.u[v] = *(const unsigned int*)&L[row][kb];
        }
    };

    const int iters = Kd >> 5;
    loadRegs();
    storeLds(0);
    __syncthreads();
    for (int it = 0; it < iters; ++it) {
        const int buf  = it & 1;
        const bool more = (it + 1 < iters);
        if (more) loadRegs();                // next tile's VMEM overlaps compute

        FragAB fa0, fa1;
        loadFrag(lds_a[buf], wm * 32 +      l16, fa0);
        loadFrag(lds_a[buf], wm * 32 + 16 + l16, fa1);
#pragma unroll
        for (int t = 0; t < 4; ++t) {
            FragAB fb;
            loadFrag(lds_b[buf], wn * 64 + t * 16 + l16, fb);
            acc[0][t] = __builtin_amdgcn_wmma_f32_16x16x32_bf16(
                            false, fa0.bf, false, fb.bf, (short)0, acc[0][t], false, false);
            acc[1][t] = __builtin_amdgcn_wmma_f32_16x16x32_bf16(
                            false, fa1.bf, false, fb.bf, (short)0, acc[1][t], false, false);
        }
        if (more) storeLds(buf ^ 1);         // fill other buffer, no read conflict
        __syncthreads();                     // single barrier per K-step
    }

    // --- epilogue: C layout VGPR r -> M = r + 8*half, lane%16 -> N ---
#pragma unroll
    for (int mi = 0; mi < 2; ++mi)
#pragma unroll
        for (int t = 0; t < 4; ++t) {
            const int n  = n0 + wn * 64 + t * 16 + l16;
            const float bv = bias ? bias[n] : 0.f;
#pragma unroll
            for (int r = 0; r < 8; ++r) {
                const int m = m0 + wm * 32 + mi * 16 + half * 8 + r;
                const float val = acc[mi][t][r] * alpha + bv;
                if constexpr (OUTF32)
                    ((float*)Cptr)[(size_t)z * sC + (size_t)m * ldc + n] = val;
                else
                    ((unsigned short*)Cptr)[(size_t)z * sC + (size_t)m * ldc + n] =
                        f32_to_bf16_rne(val);
            }
        }
}

// ---------------------------------------------------------------------------
// Weight prep: W [K,N] f32 -> WT [N,K] bf16 (transposed, converted once/call)
// ---------------------------------------------------------------------------
__global__ __launch_bounds__(256)
void transpose_f32_bf16(const float* __restrict__ src, unsigned short* __restrict__ dst,
                        int K, int N) {
    __shared__ float t[32][33];
    const int n0 = blockIdx.x * 32, k0 = blockIdx.y * 32;
    const int tx = threadIdx.x & 31, ty = threadIdx.x >> 5;   // 32 x 8
    for (int i = ty; i < 32; i += 8)
        t[i][tx] = src[(size_t)(k0 + i) * N + (n0 + tx)];
    __syncthreads();
    for (int i = ty; i < 32; i += 8)
        dst[(size_t)(n0 + i) * K + (k0 + tx)] = f32_to_bf16_rne(t[tx][i]);
}

// ---------------------------------------------------------------------------
// Embedding gather straight to bf16
// ---------------------------------------------------------------------------
__global__ void embed_gather_bf16(const float* __restrict__ emb, const int* __restrict__ idx,
                                  unsigned short* __restrict__ out, int D) {
    const int bs = blockIdx.x;
    const float* src = emb + (size_t)idx[bs] * D;
    unsigned short* dst = out + (size_t)bs * D;
    for (int d = threadIdx.x; d < D; d += blockDim.x)
        dst[d] = f32_to_bf16_rne(src[d]);
}

// ---------------------------------------------------------------------------
// Row softmax: f32 scores [B*S, cols] -> bf16 probs; optional causal mask.
// ---------------------------------------------------------------------------
__global__ __launch_bounds__(256)
void softmax_rows_bf16(const float* __restrict__ S_, unsigned short* __restrict__ P,
                       int cols, int seq, int causal) {
    __shared__ float red[256];
    const int row = blockIdx.x;
    const int q   = row % seq;
    const int len = causal ? (q + 1) : cols;
    const float* r = S_ + (size_t)row * cols;
    unsigned short* p = P + (size_t)row * cols;
    const int tid = threadIdx.x;

    float lmax = -3.4e38f;
    for (int c = tid; c < len; c += 256) lmax = fmaxf(lmax, r[c]);
    red[tid] = lmax;
    __syncthreads();
    for (int s = 128; s > 0; s >>= 1) {
        if (tid < s) red[tid] = fmaxf(red[tid], red[tid + s]);
        __syncthreads();
    }
    const float m = red[0];
    __syncthreads();

    float lsum = 0.f;
    for (int c = tid; c < len; c += 256) lsum += __expf(r[c] - m);
    red[tid] = lsum;
    __syncthreads();
    for (int s = 128; s > 0; s >>= 1) {
        if (tid < s) red[tid] += red[tid + s];
        __syncthreads();
    }
    const float inv = 1.f / red[0];

    for (int c = tid; c < len; c += 256) p[c] = f32_to_bf16_rne(__expf(r[c] - m) * inv);
    for (int c = len + tid; c < cols; c += 256) p[c] = 0;   // masked cols
}

// ---------------------------------------------------------------------------
// Host orchestration
// ---------------------------------------------------------------------------
namespace {
constexpr int kB = 4, kS = 1024, kD = 1024, kV = 32000;
constexpr int kTok = kB * kS;

static void gemm(hipStream_t s, const unsigned short* A, const unsigned short* B,
                 const float* bias, void* C, int M, int N, int Kd,
                 int lda, int ldb, int ldc,
                 long long sA, long long sB, long long sC, int batch,
                 float alpha, int transB, int outF32) {
    dim3 g(N / 128, M / 128, batch), b(256);
    if (transB) {
        if (outF32)
            hipLaunchKernelGGL((gemm_bf16_wmma<1, 1>), g, b, 0, s, A, B, bias, C,
                               Kd, lda, ldb, ldc, sA, sB, sC, alpha);
        else
            hipLaunchKernelGGL((gemm_bf16_wmma<1, 0>), g, b, 0, s, A, B, bias, C,
                               Kd, lda, ldb, ldc, sA, sB, sC, alpha);
    } else {
        if (outF32)
            hipLaunchKernelGGL((gemm_bf16_wmma<0, 1>), g, b, 0, s, A, B, bias, C,
                               Kd, lda, ldb, ldc, sA, sB, sC, alpha);
        else
            hipLaunchKernelGGL((gemm_bf16_wmma<0, 0>), g, b, 0, s, A, B, bias, C,
                               Kd, lda, ldb, ldc, sA, sB, sC, alpha);
    }
}
}  // namespace

extern "C" void kernel_launch(void* const* d_in, const int* in_sizes, int n_in,
                              void* d_out, int out_size, void* d_ws, size_t ws_size,
                              hipStream_t stream) {
    (void)in_sizes; (void)n_in; (void)out_size; (void)ws_size;

    const int*   x_enc = (const int*)d_in[0];
    const int*   x_dec = (const int*)d_in[1];
    /* d_in[2] = mascara_causal (tril) -- applied analytically in softmax */
    const float* emb   = (const float*)d_in[3];
    const float* W[9]; const float* bW[9];
    for (int i = 0; i < 9; ++i) {
        W[i]  = (const float*)d_in[4 + 2 * i];
        bW[i] = (const float*)d_in[5 + 2 * i];
    }
    const float* W_out = (const float*)d_in[22];
    const float* b_out = (const float*)d_in[23];
    float* out = (float*)d_out;

    // ---- workspace layout (bytes) ----
    char* w = (char*)d_ws;
    const size_t ACT2 = (size_t)kTok * kD * 2;        // 8 MB  (bf16 activations)
    const size_t SS2  = (size_t)kB * kS * kS * 2;     // 8 MB  (bf16 probs)
    const size_t SS4  = (size_t)kB * kS * kS * 4;     // 16 MB (f32 scores)
    const size_t WT2  = (size_t)kD * kD * 2;          // 2 MB  per weight
    unsigned short* Xb = (unsigned short*)(w);
    unsigned short* Yb = (unsigned short*)(w + 1 * ACT2);
    unsigned short* Qb = (unsigned short*)(w + 2 * ACT2);
    unsigned short* Kb = (unsigned short*)(w + 3 * ACT2);
    unsigned short* Vb = (unsigned short*)(w + 4 * ACT2);
    unsigned short* Zb = (unsigned short*)(w + 5 * ACT2);
    unsigned short* Ab = (unsigned short*)(w + 6 * ACT2);
    unsigned short* Pb = (unsigned short*)(w + 7 * ACT2);
    float*          SC = (float*)(w + 7 * ACT2 + SS2);
    unsigned short* WT[9];
    char* wtb = w + 7 * ACT2 + SS2 + SS4;
    for (int i = 0; i < 9; ++i) WT[i] = (unsigned short*)(wtb + (size_t)i * WT2);
    unsigned short* WoT = (unsigned short*)(wtb + 9 * WT2);   // [V, D] bf16, 64 MB

    // ---- weight prep: transpose+convert to bf16 [N,K] ----
    for (int i = 0; i < 9; ++i)
        hipLaunchKernelGGL(transpose_f32_bf16, dim3(kD / 32, kD / 32), dim3(256), 0, stream,
                           W[i], WT[i], kD, kD);
    hipLaunchKernelGGL(transpose_f32_bf16, dim3(kV / 32, kD / 32), dim3(256), 0, stream,
                       W_out, WoT, kD, kV);

    // ---- embeddings (gather -> bf16) ----
    hipLaunchKernelGGL(embed_gather_bf16, dim3(kTok), dim3(256), 0, stream, emb, x_enc, Xb, kD);
    hipLaunchKernelGGL(embed_gather_bf16, dim3(kTok), dim3(256), 0, stream, emb, x_dec, Yb, kD);

    const long long sQK = (long long)kS * kD;
    const long long sSS = (long long)kS * kS;
    const float inv_sqrt_d = 0.03125f;   // 1/sqrt(1024)

    // ---- encoder self-attention -> Zb ----
    gemm(stream, Xb, WT[0], bW[0], Qb, kTok, kD, kD, kD, kD, kD, 0, 0, 0, 1, 1.f, 1, 0);
    gemm(stream, Xb, WT[1], bW[1], Kb, kTok, kD, kD, kD, kD, kD, 0, 0, 0, 1, 1.f, 1, 0);
    gemm(stream, Xb, WT[2], bW[2], Vb, kTok, kD, kD, kD, kD, kD, 0, 0, 0, 1, 1.f, 1, 0);
    gemm(stream, Qb, Kb, nullptr, SC, kS, kS, kD, kD, kD, kS, sQK, sQK, sSS, kB, inv_sqrt_d, 1, 1);
    hipLaunchKernelGGL(softmax_rows_bf16, dim3(kTok), dim3(256), 0, stream, SC, Pb, kS, kS, 0);
    gemm(stream, Pb, Vb, nullptr, Zb, kS, kD, kS, kS, kD, kD, sSS, sQK, sQK, kB, 1.f, 0, 0);

    // ---- decoder masked self-attention -> Ab ----
    gemm(stream, Yb, WT[3], bW[3], Qb, kTok, kD, kD, kD, kD, kD, 0, 0, 0, 1, 1.f, 1, 0);
    gemm(stream, Yb, WT[4], bW[4], Kb, kTok, kD, kD, kD, kD, kD, 0, 0, 0, 1, 1.f, 1, 0);
    gemm(stream, Yb, WT[5], bW[5], Vb, kTok, kD, kD, kD, kD, kD, 0, 0, 0, 1, 1.f, 1, 0);
    gemm(stream, Qb, Kb, nullptr, SC, kS, kS, kD, kD, kD, kS, sQK, sQK, sSS, kB, inv_sqrt_d, 1, 1);
    hipLaunchKernelGGL(softmax_rows_bf16, dim3(kTok), dim3(256), 0, stream, SC, Pb, kS, kS, 1);
    gemm(stream, Pb, Vb, nullptr, Ab, kS, kD, kS, kS, kD, kD, sSS, sQK, sQK, kB, 1.f, 0, 0);

    // ---- cross-attention: Q from Ab, K/V from Zb -> Ab ----
    gemm(stream, Ab, WT[6], bW[6], Qb, kTok, kD, kD, kD, kD, kD, 0, 0, 0, 1, 1.f, 1, 0);
    gemm(stream, Zb, WT[7], bW[7], Kb, kTok, kD, kD, kD, kD, kD, 0, 0, 0, 1, 1.f, 1, 0);
    gemm(stream, Zb, WT[8], bW[8], Vb, kTok, kD, kD, kD, kD, kD, 0, 0, 0, 1, 1.f, 1, 0);
    gemm(stream, Qb, Kb, nullptr, SC, kS, kS, kD, kD, kD, kS, sQK, sQK, sSS, kB, inv_sqrt_d, 1, 1);
    hipLaunchKernelGGL(softmax_rows_bf16, dim3(kTok), dim3(256), 0, stream, SC, Pb, kS, kS, 0);
    gemm(stream, Pb, Vb, nullptr, Ab, kS, kD, kS, kS, kD, kD, sSS, sQK, sQK, kB, 1.f, 0, 0);

    // ---- vocab projection (dominant: 268 GFLOP, bf16 W_outT streams from L2) ----
    gemm(stream, Ab, WoT, b_out, out, kTok, kV, kD, kD, kD, kV, 0, 0, 0, 1, 1.f, 1, 1);
}
